// ReplacedWordLMHeadModel_12713103196255
// MI455X (gfx1250) — matrix operand order repaired and checked by reading, not compiled, not run on hardware
//
#include <hip/hip_runtime.h>
#include <hip/hip_bf16.h>

// ---------------------------------------------------------------------------
// ReplacedWordLMHeadModel for MI455X (gfx1250, wave32, WMMA).
//
// Shapes: B=2, K=16, S=1024, D=768, V=50257, NSENSE=4096.
// Strategy: cast to BF16, run both GEMMs on V_WMMA_F32_16X16X32_BF16 with
// f32 accumulation. Sense-table replacement fused into GEMM1 B staging.
// GEMM2 stages its (already-bf16) tiles with GLOBAL_LOAD_ASYNC_TO_LDS_B128
// (ASYNCcnt path, CDNA5 ISA).
// ---------------------------------------------------------------------------

#define BB      2
#define KH      16
#define SS      1024
#define DD      768
#define VOCAB   50257
#define NSENSE  4096
#define KSPLIT  4

typedef __attribute__((ext_vector_type(16))) __bf16   v16bf;
typedef __attribute__((ext_vector_type(8)))  float    v8f;
typedef __attribute__((ext_vector_type(4)))  unsigned v4u;
typedef __attribute__((ext_vector_type(2)))  float    v2f;

union U16 { v16bf v; v4u q[2]; };

// ---- async global->LDS staging (gfx1250 ASYNCcnt path), guarded ----------
#if defined(__gfx1250__) && __has_builtin(__builtin_amdgcn_global_load_async_to_lds_b128)
#define ATHENA_ASYNC_LDS 1
// builtin signature (per hipcc diagnostic): pointee is vector_size(16) int,
// arg0 in addrspace(1) (global), arg1 in addrspace(3) (LDS), then imm offset
// and imm cpol.
typedef int v4i_vs __attribute__((vector_size(16)));
typedef __attribute__((address_space(1))) v4i_vs gas_v4i;
typedef __attribute__((address_space(3))) v4i_vs las_v4i;
__device__ __forceinline__ void async_copy_b128(const void* g, const void* l) {
    // per-lane: LDS[l .. l+15] = MEM[g .. g+15], tracked with ASYNCcnt.
    // Generic LDS address truncated to 32 bits == raw LDS offset (aperture
    // mapping: addr[31:0] = LDS offset).
    __builtin_amdgcn_global_load_async_to_lds_b128(
        (gas_v4i*)(uintptr_t)g, (las_v4i*)(unsigned)(uintptr_t)l, 0, 0);
}
__device__ __forceinline__ void wait_async0() {
#if __has_builtin(__builtin_amdgcn_s_wait_asynccnt)
    __builtin_amdgcn_s_wait_asynccnt(0);
#else
    asm volatile("s_wait_asynccnt 0" ::: "memory");
#endif
}
#else
#define ATHENA_ASYNC_LDS 0
#endif

// round-to-nearest-even f32 -> bf16, packed pair into one dword
__device__ __forceinline__ unsigned pack_bf16(float x, float y) {
    unsigned bx = __float_as_uint(x);
    unsigned by = __float_as_uint(y);
    bx = (bx + 0x7FFFu + ((bx >> 16) & 1u)) >> 16;
    by = (by + 0x7FFFu + ((by >> 16) & 1u)) >> 16;
    return (bx & 0xFFFFu) | (by << 16);
}

// LDS tile row stride in dwords: 16 data dwords (32 bf16) padded to 20
// (80 B = multiple of 16 B so v4u reads stay aligned; conflict-free operand
// fetch pattern across 64 banks).
#define LROW 20

// ---------------------------------------------------------------------------
// Kernel 1: lm_w f32 -> bf16 (packed dwords), done once; L2-resident after.
// ---------------------------------------------------------------------------
__global__ __launch_bounds__(256) void convert_lmw_kernel(
    const float* __restrict__ w, unsigned* __restrict__ wb, int ndw) {
    int i = blockIdx.x * 256 + threadIdx.x;
    if (i < ndw) {
        v2f f = *(const v2f*)(w + 2 * (size_t)i);
        wb[i] = pack_bf16(f.x, f.y);
    }
}

// ---------------------------------------------------------------------------
// Shared WMMA tile engine: one 256-thread block computes a 128x128 tile.
// 8 waves arranged 4(M) x 2(N); each wave: 2x4 grid of 16x16 f32 accumulators.
// sA: A-tile 128(M) x 32(Kc) bf16, row-major.  sB: B-tile stored TRANSPOSED
// as [128(N)][32(Kc)] bf16 so B-operand fetch is contiguous along K.
// Operand->VGPR packing follows CDNA5 ISA 16-bit layouts.
// ---------------------------------------------------------------------------
__device__ __forceinline__ void wmma_tile_step(
    const unsigned* __restrict__ sA, const unsigned* __restrict__ sB,
    int lane, int wm, int wn, v8f acc[2][4]) {
    U16 a0, a1, bb[4];
    const int aoff = (lane >> 4) ? 4 : 0;   // lanes>=16 take K=8..15 / 24..31
    const unsigned* rowA0 = &sA[(32 * wm + (lane & 15)) * LROW];
    const unsigned* rowA1 = rowA0 + 16 * LROW;
    a0.q[0] = *(const v4u*)(rowA0 + aoff);
    a0.q[1] = *(const v4u*)(rowA0 + aoff + 8);
    a1.q[0] = *(const v4u*)(rowA1 + aoff);
    a1.q[1] = *(const v4u*)(rowA1 + aoff + 8);
    const int boff = (lane >> 4) ? 4 : 0;   // lanes>=16 take K=8..15 / 24..31
#pragma unroll
    for (int c = 0; c < 4; ++c) {
        const unsigned* rowB = &sB[(64 * wn + 16 * c + (lane & 15)) * LROW];
        bb[c].q[0] = *(const v4u*)(rowB + boff);
        bb[c].q[1] = *(const v4u*)(rowB + boff + 8);
    }
#pragma unroll
    for (int c = 0; c < 4; ++c) {
        acc[0][c] = __builtin_amdgcn_wmma_f32_16x16x32_bf16(
            false, a0.v, false, bb[c].v, (short)0, acc[0][c], false, false);
        acc[1][c] = __builtin_amdgcn_wmma_f32_16x16x32_bf16(
            false, a1.v, false, bb[c].v, (short)0, acc[1][c], false, false);
    }
}

// ---------------------------------------------------------------------------
// Kernel 2: GEMM1. hidden_part[ks][b][s][d] = sum over this split's k of
//   ctx[b,k,s,:] @ content'[b,k,:,d], with content' = sense-replaced content.
// grid = (D/128=6, S/128=8, B*KSPLIT=8), block = 256.
// ---------------------------------------------------------------------------
__global__ __launch_bounds__(256) void gemm1_kernel(
    const int*   __restrict__ input_ids,   // [B][S]
    const float* __restrict__ ctx,         // [B][K][S][S]
    const float* __restrict__ content,     // [B][K][S][D]
    const float* __restrict__ sense,       // [NSENSE][K][D]
    float*       __restrict__ hidden_part) // [KSPLIT][B][S][D]
{
    __shared__ int s_ids[SS];
    __shared__ alignas(16) unsigned sA[128 * LROW];
    __shared__ alignas(16) unsigned sB[128 * LROW];

    const int tid  = threadIdx.x;
    const int lane = tid & 31;
    const int wave = tid >> 5;
    const int wm   = wave & 3;
    const int wn   = wave >> 2;
    const int n0   = blockIdx.x * 128;          // D tile
    const int m0   = blockIdx.y * 128;          // S tile
    const int b    = blockIdx.z >> 2;
    const int ks   = blockIdx.z & (KSPLIT - 1);

    // sense-dictionary membership staged once per block
    for (int i = tid; i < SS; i += 256) {
        int id = input_ids[b * SS + i];
        s_ids[i] = (id < NSENSE) ? id : -1;
    }

    const v8f zf = {0.f, 0.f, 0.f, 0.f, 0.f, 0.f, 0.f, 0.f};
    v8f acc[2][4];
#pragma unroll
    for (int g = 0; g < 2; ++g)
#pragma unroll
        for (int c = 0; c < 4; ++c) acc[g][c] = zf;

    const int kbeg = ks * (KH / KSPLIT);
    const int kend = kbeg + (KH / KSPLIT);
    for (int k = kbeg; k < kend; ++k) {
        const float* ctxk   = ctx + (((size_t)b * KH + k) * SS + m0) * SS;
        const float* contk  = content + ((size_t)b * KH + k) * SS * DD;
        const float* sensek = sense + (size_t)k * DD;
        for (int t0 = 0; t0 < SS; t0 += 32) {
            __syncthreads();
            // ---- stage A: 128 x 32 of ctx (f32 -> packed bf16) ----
            for (int idx = tid; idx < 128 * 16; idx += 256) {
                int r = idx >> 4, c = idx & 15;
                v2f f = *(const v2f*)(ctxk + (size_t)r * SS + t0 + c * 2);
                sA[r * LROW + c] = pack_bf16(f.x, f.y);
            }
            // prefetch next A tile into L2/L0 (global_prefetch_b8)
            if (t0 + 32 < SS)
                __builtin_prefetch(ctxk + (size_t)(tid & 127) * SS + t0 + 32, 0, 1);
            // ---- stage B transposed: sB[d][t], fusing sense replacement ----
            for (int idx = tid; idx < 128 * 16; idx += 256) {
                int d  = idx & 127;                // coalesced along d
                int tp = idx >> 7;                 // 0..15 (t pairs)
                int ta = t0 + tp * 2, tb = ta + 1;
                int ia = s_ids[ta], ib = s_ids[tb];
                float fa = (ia >= 0) ? sensek[(size_t)ia * (KH * DD) + n0 + d]
                                     : contk[(size_t)ta * DD + n0 + d];
                float fb = (ib >= 0) ? sensek[(size_t)ib * (KH * DD) + n0 + d]
                                     : contk[(size_t)tb * DD + n0 + d];
                sB[d * LROW + tp] = pack_bf16(fa, fb);
            }
            __syncthreads();
            wmma_tile_step(sA, sB, lane, wm, wn, acc);
        }
    }

    // ---- store f32 partial (disjoint region per (ks,b,tile): no atomics) ----
    float* outp = hidden_part + ((size_t)ks * BB + b) * SS * DD;
    const int mrow = m0 + 32 * wm + ((lane >> 4) ? 8 : 0);
    const int ncol = n0 + 64 * wn + (lane & 15);
#pragma unroll
    for (int g = 0; g < 2; ++g)
#pragma unroll
        for (int c = 0; c < 4; ++c)
#pragma unroll
            for (int r = 0; r < 8; ++r)
                outp[(size_t)(mrow + 16 * g + r) * DD + (ncol + 16 * c)] =
                    acc[g][c][r];
}

// ---------------------------------------------------------------------------
// Kernel 3: reduce the KSPLIT partials and convert hidden -> bf16.
// ---------------------------------------------------------------------------
__global__ __launch_bounds__(256) void reduce_hidden_kernel(
    const float* __restrict__ hp, unsigned* __restrict__ hb) {
    const int i = blockIdx.x * 256 + threadIdx.x;   // dword index
    const int NTOT = BB * SS * DD;
    const int e = i * 2;
    float x = 0.f, y = 0.f;
#pragma unroll
    for (int ks = 0; ks < KSPLIT; ++ks) {
        x += hp[(size_t)ks * NTOT + e];
        y += hp[(size_t)ks * NTOT + e + 1];
    }
    hb[i] = pack_bf16(x, y);
}

// ---------------------------------------------------------------------------
// Kernel 4: GEMM2. logits[m][v] = hidden_bf16[m,:] . lm_w_bf16[v,:]
// grid = (ceil(V/128)=393, (B*S)/128=16), block = 256.
// Tiles are already bf16 in global memory -> staging is a pure copy, done
// with GLOBAL_LOAD_ASYNC_TO_LDS_B128 (no VGPR round trip).
// ---------------------------------------------------------------------------
__global__ __launch_bounds__(256) void gemm2_kernel(
    const unsigned* __restrict__ hb,   // hidden bf16 dwords [2048][384]
    const unsigned* __restrict__ wb,   // lm_w  bf16 dwords [VOCAB][384]
    float*          __restrict__ out)  // [2048][VOCAB]
{
    __shared__ alignas(16) unsigned sA[128 * LROW];
    __shared__ alignas(16) unsigned sB[128 * LROW];

    const int tid  = threadIdx.x;
    const int lane = tid & 31;
    const int wave = tid >> 5;
    const int wm   = wave & 3;
    const int wn   = wave >> 2;
    const int n0   = blockIdx.x * 128;   // vocab tile (last one partial)
    const int m0   = blockIdx.y * 128;   // token tile

    const v8f zf = {0.f, 0.f, 0.f, 0.f, 0.f, 0.f, 0.f, 0.f};
    v8f acc[2][4];
#pragma unroll
    for (int g = 0; g < 2; ++g)
#pragma unroll
        for (int c = 0; c < 4; ++c) acc[g][c] = zf;

    for (int k0 = 0; k0 < DD; k0 += 32) {
        const int kd = k0 >> 1;          // dword column base (16 dwords/row)
        __syncthreads();
#if ATHENA_ASYNC_LDS
        // A tile: 128 rows x 16 dwords = 4 b128 chunks per row (all 16B aligned)
        for (int idx = tid; idx < 128 * 4; idx += 256) {
            int r = idx >> 2, c = (idx & 3) * 4;
            async_copy_b128(&hb[(size_t)(m0 + r) * (DD / 2) + kd + c],
                            &sA[r * LROW + c]);
        }
        // B tile: same pattern with vocab-edge zero fill
        for (int idx = tid; idx < 128 * 4; idx += 256) {
            int r = idx >> 2, c = (idx & 3) * 4;
            int v = n0 + r;
            if (v < VOCAB) {
                async_copy_b128(&wb[(size_t)v * (DD / 2) + kd + c],
                                &sB[r * LROW + c]);
            } else {
                v4u z = {0u, 0u, 0u, 0u};
                *(v4u*)&sB[r * LROW + c] = z;
            }
        }
        wait_async0();
#else
        for (int idx = tid; idx < 128 * 16; idx += 256) {
            int r = idx >> 4, c = idx & 15;
            sA[r * LROW + c] = hb[(size_t)(m0 + r) * (DD / 2) + kd + c];
        }
        for (int idx = tid; idx < 128 * 16; idx += 256) {
            int r = idx >> 4, c = idx & 15;
            int v = n0 + r;
            sB[r * LROW + c] =
                (v < VOCAB) ? wb[(size_t)v * (DD / 2) + kd + c] : 0u;
        }
#endif
        __syncthreads();
        wmma_tile_step(sA, sB, lane, wm, wn, acc);
    }

    const int mrow = m0 + 32 * wm + ((lane >> 4) ? 8 : 0);
    const int nbase = n0 + 64 * wn + (lane & 15);
#pragma unroll
    for (int g = 0; g < 2; ++g)
#pragma unroll
        for (int c = 0; c < 4; ++c) {
            int col = nbase + 16 * c;
            if (col < VOCAB) {
#pragma unroll
                for (int r = 0; r < 8; ++r)
                    out[(size_t)(mrow + 16 * g + r) * VOCAB + col] =
                        acc[g][c][r];
            }
        }
}

// ---------------------------------------------------------------------------
// Launch. Workspace layout (bytes):
//   [0, 25165824)              hidden_part f32 [KSPLIT][B][S][D]
//   [25165824, 28311552)       hidden_bf16 dwords [B*S][D/2]
//   [28311552, 105506304)      lm_w_bf16 dwords [VOCAB][D/2]
// ---------------------------------------------------------------------------
extern "C" void kernel_launch(void* const* d_in, const int* in_sizes, int n_in,
                              void* d_out, int out_size, void* d_ws, size_t ws_size,
                              hipStream_t stream) {
    const int*   input_ids = (const int*)  d_in[0];
    const float* ctx       = (const float*)d_in[1];
    const float* content   = (const float*)d_in[2];
    const float* sense     = (const float*)d_in[3];
    const float* lmw       = (const float*)d_in[4];
    // d_in[5] = n_sense_words (== NSENSE, compile-time constant here)

    char* ws = (char*)d_ws;
    float*    hidden_part = (float*)ws;
    unsigned* hidden_bf16 = (unsigned*)(ws + (size_t)KSPLIT * BB * SS * DD * 4);
    unsigned* lmw_bf16    = (unsigned*)(ws + (size_t)KSPLIT * BB * SS * DD * 4
                                           + (size_t)BB * SS * (DD / 2) * 4);
    float* logits = (float*)d_out;

    const int lmw_ndw = VOCAB * (DD / 2);
    convert_lmw_kernel<<<(lmw_ndw + 255) / 256, 256, 0, stream>>>(
        lmw, lmw_bf16, lmw_ndw);

    gemm1_kernel<<<dim3(DD / 128, SS / 128, BB * KSPLIT), 256, 0, stream>>>(
        input_ids, ctx, content, sense, hidden_part);

    reduce_hidden_kernel<<<(BB * SS * DD / 2) / 256, 256, 0, stream>>>(
        hidden_part, hidden_bf16);

    gemm2_kernel<<<dim3((VOCAB + 127) / 128, (BB * SS) / 128), 256, 0, stream>>>(
        hidden_bf16, lmw_bf16, logits);
}
